// MAGBlock_89361089561462
// MI455X (gfx1250) — matrix operand order repaired and checked by reading, not compiled, not run on hardware
//
#include <hip/hip_runtime.h>

// ---------------- types ----------------
typedef __bf16 bf16;
typedef __attribute__((ext_vector_type(16))) __bf16 v16bf;
typedef __attribute__((ext_vector_type(8)))  __bf16 v8bf;
typedef __attribute__((ext_vector_type(8)))  float  v8f;
typedef __attribute__((ext_vector_type(4)))  unsigned int u32x4;
typedef __attribute__((ext_vector_type(8)))  int i32x8;
typedef __attribute__((ext_vector_type(4)))  int i32x4;

union BF16x16 { v16bf v; v8bf h[2]; };

__device__ __forceinline__ v8f wmma_bf16(const BF16x16& a, const BF16x16& b, v8f c) {
  // D = A(16x32 bf16) * B(32x16 bf16) + C(16x16 f32), wave32
  return __builtin_amdgcn_wmma_f32_16x16x32_bf16(false, a.v, false, b.v,
                                                 (short)0, c, false, false);
}

__device__ __forceinline__ float sigmoidf_(float x) { return 1.f / (1.f + __expf(-x)); }
__device__ __forceinline__ float siluf_(float x)    { return x / (1.f + __expf(-x)); }

// compiler-only barrier: LDS contents may have been changed by the TDM engine
#define MEM_CLOBBER() asm volatile("" ::: "memory")

// wait for all outstanding TDM transfers; memory clobber so the compiler
// cannot fold away LDS reads of TDM-written data (TDM writes are not visible
// to alias analysis as stores).
__device__ __forceinline__ void tdm_wait0() {
  asm volatile("s_wait_tensorcnt 0x0" ::: "memory");
}

// ---------------- constants ----------------
#define SEQ   2048
#define DIM   2048
#define NH    16
#define DH    128
#define N3    6144      // 3*DIM
#define MH    8192      // mem hidden
#define FH    5461      // ffn hidden (true)
#define FHP   5504      // ffn hidden padded: 43*128, mult of BK and BN
#define EPSN  1e-6f
#define LOGDECAY (-0.105360515657826f)   // ln(0.9)

// ================= Tensor Data Mover: 2D tile global -> LDS =================
// D# built per CDNA5 ISA (group0 + group1; groups 2/3 zero).
// tile: tile_rows x tile_cols elements (2-byte), row stride = tensor_cols elems.
// This toolchain exposes the 6-arg builtin:
//   (uint32x4 g0, int32x8 g1, int32x4 g2, int32x4 g3, int32x8 pad, i32 cpol)
__device__ __forceinline__ void tdm_load_2d(unsigned lds_byte_off, const void* gptr,
                                            unsigned tile_rows, unsigned tile_cols,
                                            unsigned tensor_cols, unsigned tensor_rows) {
  unsigned long long ga = (unsigned long long)(uintptr_t)gptr;
  u32x4 g0;
  g0.x = 1u;                                          // count=1, user mode
  g0.y = lds_byte_off;                                // LDS dest (bytes)
  g0.z = (unsigned)(ga & 0xFFFFFFFFu);                // global_addr[31:0]
  g0.w = (unsigned)((ga >> 32) & 0x01FFFFFFu) | (2u << 30);  // addr[56:32] | type=2
  i32x8 g1;
  g1[0] = (int)(1u << 16);                                       // data_size=1 (2B)
  g1[1] = (int)((tensor_cols & 0xFFFFu) << 16);                  // tensor_dim0[15:0]
  g1[2] = (int)((tensor_cols >> 16) | ((tensor_rows & 0xFFFFu) << 16)); // dim0 hi | dim1 lo
  g1[3] = (int)((tensor_rows >> 16) | (tile_cols << 16));        // dim1 hi | tile_dim0
  g1[4] = (int)tile_rows;                                        // tile_dim1 (tile_dim2=0)
  g1[5] = (int)tensor_cols;                                      // tensor_dim0_stride lo
  g1[6] = 0;
  g1[7] = 0;
  i32x4 z4 = {0, 0, 0, 0};
  i32x8 z8 = {0, 0, 0, 0, 0, 0, 0, 0};
  __builtin_amdgcn_tensor_load_to_lds(g0, g1, z4, z4, z8, 0);
}

// ================= cast / pad fp32 -> bf16 =================
__global__ void k_cast_pad(const float* __restrict__ src, bf16* __restrict__ dst,
                           int src_rows, int src_cols, int dst_rows, int dst_cols) {
  size_t n = (size_t)dst_rows * dst_cols;
  for (size_t i = (size_t)blockIdx.x * blockDim.x + threadIdx.x; i < n;
       i += (size_t)gridDim.x * blockDim.x) {
    int r = (int)(i / dst_cols), c = (int)(i % dst_cols);
    float v = 0.f;
    if (r < src_rows && c < src_cols) v = src[(size_t)r * src_cols + c];
    dst[i] = (bf16)v;
  }
}

// ====== cast + transpose + pad: src [K_src x N_src] f32 -> dst [N_dst x K_dst] bf16 ======
__global__ void k_cast_pad_t(const float* __restrict__ src, bf16* __restrict__ dst,
                             int src_rows, int src_cols, int dst_rows, int dst_cols) {
  size_t n = (size_t)dst_rows * dst_cols;
  for (size_t i = (size_t)blockIdx.x * blockDim.x + threadIdx.x; i < n;
       i += (size_t)gridDim.x * blockDim.x) {
    int nn = (int)(i / dst_cols), kk = (int)(i % dst_cols);
    float v = 0.f;
    if (kk < src_rows && nn < src_cols) v = src[(size_t)kk * src_cols + nn];
    dst[i] = (bf16)v;
  }
}

// ================= RMSNorm (f32 in, bf16 out), one block per row =================
__global__ void k_rmsnorm_bf16(const float* __restrict__ x, const float* __restrict__ w,
                               bf16* __restrict__ out) {
  int s = blockIdx.x, tid = threadIdx.x;           // block 256, D=2048 -> 8 per thread
  const float* row = x + (size_t)s * DIM;
  float vals[8]; float ss = 0.f;
#pragma unroll
  for (int i = 0; i < 8; ++i) { float v = row[tid + i * 256]; vals[i] = v; ss += v * v; }
#pragma unroll
  for (int off = 1; off < 32; off <<= 1) ss += __shfl_xor(ss, off, 32);
  __shared__ float red[8];
  int wave = tid >> 5, lane = tid & 31;
  if (lane == 0) red[wave] = ss;
  __syncthreads();
  float tot = 0.f;
#pragma unroll
  for (int i = 0; i < 8; ++i) tot += red[i];
  float r = rsqrtf(tot / (float)DIM + EPSN);
#pragma unroll
  for (int i = 0; i < 8; ++i)
    out[(size_t)s * DIM + tid + i * 256] = (bf16)(vals[i] * r * w[tid + i * 256]);
}

// ================= gamma gate: sigmoid(silu(h@W1)@W2), one block(64) per token ======
__global__ void k_gamma(const bf16* __restrict__ hbf, const float* __restrict__ w1,
                        const float* __restrict__ w2, float* __restrict__ gamma) {
  int s = blockIdx.x, t = threadIdx.x;             // 64 threads
  const bf16* row = hbf + (size_t)s * DIM;
  float acc = 0.f;
  for (int i = 0; i < DIM; ++i) acc += (float)row[i] * w1[(size_t)i * 64 + t];
  float contrib = siluf_(acc) * w2[t];
#pragma unroll
  for (int off = 1; off < 32; off <<= 1) contrib += __shfl_xor(contrib, off, 32);
  __shared__ float r2[2];
  if ((t & 31) == 0) r2[t >> 5] = contrib;
  __syncthreads();
  if (t == 0) gamma[s] = sigmoidf_(r2[0] + r2[1]);
}

// ================= per-head QK RMSNorm + RoPE, split heads, bf16 out ================
// qkv: [S, 6144] f32.  outputs q/k/v: [H][S][DH] bf16
__global__ void k_qknorm_rope(const float* __restrict__ qkv, const float* __restrict__ qs,
                              const float* __restrict__ ks, bf16* __restrict__ qh,
                              bf16* __restrict__ kh_, bf16* __restrict__ vh) {
  int s = blockIdx.x, h = blockIdx.y, d = threadIdx.x;   // block 128
  __shared__ float qn[DH], kn[DH], red[8];
  const float* base = qkv + (size_t)s * N3 + h * DH;
  float qv = base[d], kv = base[DIM + d], vv = base[2 * DIM + d];
  float sq = qv * qv, sk = kv * kv;
#pragma unroll
  for (int off = 1; off < 32; off <<= 1) { sq += __shfl_xor(sq, off, 32); sk += __shfl_xor(sk, off, 32); }
  int wave = d >> 5, lane = d & 31;
  if (lane == 0) { red[wave] = sq; red[4 + wave] = sk; }
  __syncthreads();
  float tq = red[0] + red[1] + red[2] + red[3];
  float tk = red[4] + red[5] + red[6] + red[7];
  float rq = rsqrtf(tq / (float)DH + EPSN), rk = rsqrtf(tk / (float)DH + EPSN);
  qn[d] = qv * rq * qs[d];
  kn[d] = kv * rk * ks[d];
  __syncthreads();
  int f = d & 63;
  float inv_freq = __powf(10000.f, -(2.f * (float)f) / (float)DH);
  float ang = (float)s * inv_freq;
  float c = __cosf(ang), sn = __sinf(ang);
  float qrh = (d < 64) ? -qn[d + 64] : qn[d - 64];
  float krh = (d < 64) ? -kn[d + 64] : kn[d - 64];
  size_t o = ((size_t)h * SEQ + s) * DH + d;
  qh[o] = (bf16)(qn[d] * c + qrh * sn);
  kh_[o] = (bf16)(kn[d] * c + krh * sn);
  vh[o] = (bf16)vv;
}

// ====== bf16 WMMA GEMM with TDM staging: C[M,N] f32 = A[M,K] @ Bt[N,K]^T ======
// All of M, N multiples of 128; K multiple of 32. No bounds checks anywhere.
#define BM 128
#define BN 128
#define BK 32
__global__ void k_gemm_tdm(const bf16* __restrict__ A, const bf16* __restrict__ Bt,
                           float* __restrict__ C, int M, int N, int K) {
  // double-buffered: per buffer A-tile 128x32 (8KB) + B-tile 128x32 (8KB)
  __shared__ __align__(16) bf16 smem[2 * (BM * BK + BN * BK)];   // 32 KB
  const unsigned BUF_ELE = BM * BK + BN * BK;                    // 8192 elems
  const unsigned BUF_BYTES = BUF_ELE * 2;                        // 16384 bytes

  int tid = threadIdx.x;
  int wave = tid >> 5, lane = tid & 31;
  int lh = lane >> 4, ll = lane & 15;
  int wm = (wave & 3) * 32;         // wave tile row in block
  int wn = (wave >> 2) * 64;        // wave tile col in block
  int block_m = blockIdx.y * BM, block_n = blockIdx.x * BN;

  v8f zero = {};
  v8f acc[2][4];
#pragma unroll
  for (int i = 0; i < 2; ++i)
#pragma unroll
    for (int j = 0; j < 4; ++j) acc[i][j] = zero;

  int nk = K / BK;
  // prologue: DMA slab 0 into buffer 0 (wave 0 issues; TDM ignores EXEC)
  if (tid < 32) {
    tdm_load_2d(0, A + (size_t)block_m * K, BM, BK, (unsigned)K, (unsigned)M);
    tdm_load_2d(BM * BK * 2, Bt + (size_t)block_n * K, BN, BK, (unsigned)K, (unsigned)N);
    tdm_wait0();
  }
  __syncthreads();
  MEM_CLOBBER();

  for (int kt = 0; kt < nk; ++kt) {
    int buf = kt & 1;
    // kick off DMA for next slab into the other buffer
    if (kt + 1 < nk && tid < 32) {
      int k0n = (kt + 1) * BK;
      unsigned ob = (unsigned)(buf ^ 1) * BUF_BYTES;
      tdm_load_2d(ob, A + (size_t)block_m * K + k0n, BM, BK, (unsigned)K, (unsigned)M);
      tdm_load_2d(ob + BM * BK * 2, Bt + (size_t)block_n * K + k0n, BN, BK,
                  (unsigned)K, (unsigned)N);
    }
    const bf16* Asb = smem + (unsigned)buf * BUF_ELE;            // [128][32]
    const bf16* Bsb = Asb + BM * BK;                             // [128][32]
#pragma unroll
    for (int mt = 0; mt < 2; ++mt) {
      BF16x16 af;
      int am = wm + mt * 16 + ll;
      int kh8 = lh * 8;
      af.h[0] = *(const v8bf*)&Asb[am * BK + kh8];
      af.h[1] = *(const v8bf*)&Asb[am * BK + kh8 + 16];
#pragma unroll
      for (int nt = 0; nt < 4; ++nt) {
        BF16x16 bfr;
        int bn = wn + nt * 16 + ll;
        int bkh = lh * 8;
        bfr.h[0] = *(const v8bf*)&Bsb[bn * BK + bkh];
        bfr.h[1] = *(const v8bf*)&Bsb[bn * BK + bkh + 16];
        acc[mt][nt] = wmma_bf16(af, bfr, acc[mt][nt]);
      }
    }
    // wait for next-slab DMA completion, then make it visible to all waves
    if (tid < 32) tdm_wait0();
    __syncthreads();
    MEM_CLOBBER();
  }
  // ---- store (C/D layout: vgpr j -> row j+8*lh, lane ll -> col); tiles always full
#pragma unroll
  for (int mt = 0; mt < 2; ++mt)
#pragma unroll
    for (int nt = 0; nt < 4; ++nt)
#pragma unroll
      for (int j = 0; j < 8; ++j) {
        int gm = block_m + wm + mt * 16 + j + 8 * lh;
        int gn = block_n + wn + nt * 16 + ll;
        C[(size_t)gm * N + gn] = acc[mt][nt][j];
      }
}

// ================= sliding-window attention (flash, WMMA) ==========================
// grid (S/64, H), block 128 (4 waves, 16 q-rows per wave)
__global__ void k_swa(const bf16* __restrict__ qg, const bf16* __restrict__ kg,
                      const bf16* __restrict__ vg, bf16* __restrict__ attn_out) {
  __shared__ bf16 Ks[64][DH];     // 16 KB
  __shared__ bf16 VsT[DH][64];    // 16 KB  (V transposed: [dh][key])
  __shared__ bf16 Ps[64][64];     // 8 KB
  int h = blockIdx.y;
  int q0 = blockIdx.x * 64;
  int tid = threadIdx.x, wave = tid >> 5, lane = tid & 31;
  int lh = lane >> 4, ll = lane & 15, kh8 = lh * 8;
  const float scale = 0.088388347648318f;   // 128^-0.5
  const bf16* qbase = qg + (size_t)h * SEQ * DH;
  const bf16* kbase = kg + (size_t)h * SEQ * DH;
  const bf16* vbase = vg + (size_t)h * SEQ * DH;

  // Q fragments (16 rows x 128, 4 k-steps of 32)
  BF16x16 qf[4];
  int qrow = q0 + wave * 16 + ll;
#pragma unroll
  for (int ks = 0; ks < 4; ++ks) {
    qf[ks].h[0] = *(const v8bf*)&qbase[(size_t)qrow * DH + ks * 32 + kh8];
    qf[ks].h[1] = *(const v8bf*)&qbase[(size_t)qrow * DH + ks * 32 + kh8 + 16];
  }

  v8f zero = {};
  v8f Oacc[8];
#pragma unroll
  for (int t = 0; t < 8; ++t) Oacc[t] = zero;
  float mrow[8], lrow[8];
#pragma unroll
  for (int j = 0; j < 8; ++j) { mrow[j] = -1e30f; lrow[j] = 0.f; }

  int lo = q0 - 511; if (lo < 0) lo = 0;
  int kt_lo = lo >> 6, kt_hi = q0 >> 6;

  for (int kt = kt_lo; kt <= kt_hi; ++kt) {
    int k0 = kt * 64;
    // stage K row-major and V transposed
    {
      int r = tid >> 1, c0 = (tid & 1) * 64;
      const v8bf* s8 = (const v8bf*)&kbase[(size_t)(k0 + r) * DH + c0];
      v8bf* d8 = (v8bf*)&Ks[r][c0];
#pragma unroll
      for (int i = 0; i < 8; ++i) d8[i] = s8[i];
      const bf16* vs = &vbase[(size_t)(k0 + r) * DH + c0];
#pragma unroll
      for (int i = 0; i < 64; ++i) VsT[c0 + i][r] = vs[i];
      if (kt < kt_hi) {
        __builtin_prefetch(&kbase[(size_t)(k0 + 64 + r) * DH + c0], 0, 1);
        __builtin_prefetch(&vbase[(size_t)(k0 + 64 + r) * DH + c0], 0, 1);
      }
    }
    __syncthreads();
    // S = Q @ K^T  (16 x 64)
    v8f Sacc[4];
#pragma unroll
    for (int nt = 0; nt < 4; ++nt) {
      Sacc[nt] = zero;
#pragma unroll
      for (int ks = 0; ks < 4; ++ks) {
        BF16x16 bfr;
        int bn = nt * 16 + ll;             // key within tile
        bfr.h[0] = *(const v8bf*)&Ks[bn][ks * 32 + lh * 8];
        bfr.h[1] = *(const v8bf*)&Ks[bn][ks * 32 + lh * 8 + 16];
        Sacc[nt] = wmma_bf16(qf[ks], bfr, Sacc[nt]);
      }
    }
    // mask + online softmax update
    float alpha[8];
#pragma unroll
    for (int j = 0; j < 8; ++j) {
      int gi = q0 + wave * 16 + j + 8 * lh;
      float mx = -1e30f;
#pragma unroll
      for (int nt = 0; nt < 4; ++nt) {
        int gj = k0 + nt * 16 + ll;
        int rel = gi - gj;
        float sv = Sacc[nt][j] * scale;
        if (rel < 0 || rel > 511) sv = -1e30f;
        Sacc[nt][j] = sv;
        mx = fmaxf(mx, sv);
      }
#pragma unroll
      for (int off = 1; off < 16; off <<= 1) mx = fmaxf(mx, __shfl_xor(mx, off, 32));
      float mnew = fmaxf(mrow[j], mx);
      alpha[j] = __expf(mrow[j] - mnew);
      mrow[j] = mnew;
      lrow[j] *= alpha[j];
    }
#pragma unroll
    for (int t = 0; t < 8; ++t)
#pragma unroll
      for (int j = 0; j < 8; ++j) Oacc[t][j] *= alpha[j];
    // P = exp(S-m); accumulate row sums; write to LDS for next WMMA
#pragma unroll
    for (int j = 0; j < 8; ++j) {
      float psum = 0.f;
#pragma unroll
      for (int nt = 0; nt < 4; ++nt) {
        float sv = Sacc[nt][j];
        float p = (sv > -1e29f) ? __expf(sv - mrow[j]) : 0.f;
        psum += p;
        Ps[wave * 16 + j + 8 * lh][nt * 16 + ll] = (bf16)p;
      }
#pragma unroll
      for (int off = 1; off < 16; off <<= 1) psum += __shfl_xor(psum, off, 32);
      lrow[j] += psum;
    }
    __syncthreads();
    // O += P @ V  (16x64 @ 64x128)
#pragma unroll
    for (int ks2 = 0; ks2 < 2; ++ks2) {
      BF16x16 paf;
      int pr = wave * 16 + ll;
      paf.h[0] = *(const v8bf*)&Ps[pr][ks2 * 32 + kh8];
      paf.h[1] = *(const v8bf*)&Ps[pr][ks2 * 32 + kh8 + 16];
#pragma unroll
      for (int t = 0; t < 8; ++t) {
        BF16x16 bvf;
        int vn = t * 16 + ll;              // dh column
        bvf.h[0] = *(const v8bf*)&VsT[vn][ks2 * 32 + lh * 8];
        bvf.h[1] = *(const v8bf*)&VsT[vn][ks2 * 32 + lh * 8 + 16];
        Oacc[t] = wmma_bf16(paf, bvf, Oacc[t]);
      }
    }
    __syncthreads();
  }
  // normalize + store [S, D] with head-major columns
#pragma unroll
  for (int t = 0; t < 8; ++t)
#pragma unroll
    for (int j = 0; j < 8; ++j) {
      int gm = q0 + wave * 16 + j + 8 * lh;
      int gn = t * 16 + ll;
      float o = Oacc[t][j] / lrow[j];
      attn_out[(size_t)gm * DIM + h * DH + gn] = (bf16)o;
    }
}

// ================= Omega-rule memory (window 64, decay*gamma, P @ K) ================
__global__ void k_omega(const bf16* __restrict__ qg, const bf16* __restrict__ kg,
                        const float* __restrict__ gamma, bf16* __restrict__ qmem_out) {
  __shared__ bf16 Ks[64][DH];     // row-major keys
  __shared__ bf16 KsT[DH][64];    // transposed keys (for P@K B-frags)
  __shared__ bf16 Ps[64][64];
  int h = blockIdx.y;
  int q0 = blockIdx.x * 64;
  int tid = threadIdx.x, wave = tid >> 5, lane = tid & 31;
  int lh = lane >> 4, ll = lane & 15, kh8 = lh * 8;
  const float scale = 0.088388347648318f;
  const bf16* qbase = qg + (size_t)h * SEQ * DH;
  const bf16* kbase = kg + (size_t)h * SEQ * DH;

  BF16x16 qf[4];
  int qrow = q0 + wave * 16 + ll;
#pragma unroll
  for (int ks = 0; ks < 4; ++ks) {
    qf[ks].h[0] = *(const v8bf*)&qbase[(size_t)qrow * DH + ks * 32 + kh8];
    qf[ks].h[1] = *(const v8bf*)&qbase[(size_t)qrow * DH + ks * 32 + kh8 + 16];
  }

  v8f zero = {};
  v8f Oacc[8];
#pragma unroll
  for (int t = 0; t < 8; ++t) Oacc[t] = zero;
  float mrow[8], lrow[8];
#pragma unroll
  for (int j = 0; j < 8; ++j) { mrow[j] = -1e30f; lrow[j] = 0.f; }

  int lo = q0 - 63; if (lo < 0) lo = 0;
  int kt_lo = lo >> 6, kt_hi = q0 >> 6;

  for (int kt = kt_lo; kt <= kt_hi; ++kt) {
    int k0 = kt * 64;
    {
      int r = tid >> 1, c0 = (tid & 1) * 64;
      const v8bf* s8 = (const v8bf*)&kbase[(size_t)(k0 + r) * DH + c0];
      v8bf* d8 = (v8bf*)&Ks[r][c0];
#pragma unroll
      for (int i = 0; i < 8; ++i) d8[i] = s8[i];
      const bf16* src = &kbase[(size_t)(k0 + r) * DH + c0];
#pragma unroll
      for (int i = 0; i < 64; ++i) KsT[c0 + i][r] = src[i];
    }
    __syncthreads();
    v8f Sacc[4];
#pragma unroll
    for (int nt = 0; nt < 4; ++nt) {
      Sacc[nt] = zero;
#pragma unroll
      for (int ks = 0; ks < 4; ++ks) {
        BF16x16 bfr;
        int bn = nt * 16 + ll;
        bfr.h[0] = *(const v8bf*)&Ks[bn][ks * 32 + lh * 8];
        bfr.h[1] = *(const v8bf*)&Ks[bn][ks * 32 + lh * 8 + 16];
        Sacc[nt] = wmma_bf16(qf[ks], bfr, Sacc[nt]);
      }
    }
    float alpha[8];
#pragma unroll
    for (int j = 0; j < 8; ++j) {
      int gi = q0 + wave * 16 + j + 8 * lh;
      float mx = -1e30f;
#pragma unroll
      for (int nt = 0; nt < 4; ++nt) {
        int gj = k0 + nt * 16 + ll;
        int rel = gi - gj;
        float sv = Sacc[nt][j] * scale;
        if (rel < 0 || rel > 63) sv = -1e30f;
        Sacc[nt][j] = sv;
        mx = fmaxf(mx, sv);
      }
#pragma unroll
      for (int off = 1; off < 16; off <<= 1) mx = fmaxf(mx, __shfl_xor(mx, off, 32));
      float mnew = fmaxf(mrow[j], mx);
      alpha[j] = __expf(mrow[j] - mnew);
      mrow[j] = mnew;
      lrow[j] *= alpha[j];
    }
#pragma unroll
    for (int t = 0; t < 8; ++t)
#pragma unroll
      for (int j = 0; j < 8; ++j) Oacc[t][j] *= alpha[j];
#pragma unroll
    for (int j = 0; j < 8; ++j) {
      int gi = q0 + wave * 16 + j + 8 * lh;
      float esum = 0.f;
#pragma unroll
      for (int nt = 0; nt < 4; ++nt) {
        int gj = k0 + nt * 16 + ll;
        int rel = gi - gj;
        float sv = Sacc[nt][j];
        float e = (sv > -1e29f) ? __expf(sv - mrow[j]) : 0.f;
        esum += e;
        float p = e;
        if (e > 0.f) p = e * __expf(LOGDECAY * (float)rel) * gamma[gj];
        Ps[wave * 16 + j + 8 * lh][nt * 16 + ll] = (bf16)p;
      }
#pragma unroll
      for (int off = 1; off < 16; off <<= 1) esum += __shfl_xor(esum, off, 32);
      lrow[j] += esum;
    }
    __syncthreads();
    // q_mem += P @ K
#pragma unroll
    for (int ks2 = 0; ks2 < 2; ++ks2) {
      BF16x16 paf;
      int pr = wave * 16 + ll;
      paf.h[0] = *(const v8bf*)&Ps[pr][ks2 * 32 + kh8];
      paf.h[1] = *(const v8bf*)&Ps[pr][ks2 * 32 + kh8 + 16];
#pragma unroll
      for (int t = 0; t < 8; ++t) {
        BF16x16 bkf;
        int vn = t * 16 + ll;
        bkf.h[0] = *(const v8bf*)&KsT[vn][ks2 * 32 + lh * 8];
        bkf.h[1] = *(const v8bf*)&KsT[vn][ks2 * 32 + lh * 8 + 16];
        Oacc[t] = wmma_bf16(paf, bkf, Oacc[t]);
      }
    }
    __syncthreads();
  }
#pragma unroll
  for (int t = 0; t < 8; ++t)
#pragma unroll
    for (int j = 0; j < 8; ++j) {
      int gm = q0 + wave * 16 + j + 8 * lh;
      int gn = t * 16 + ll;
      float o = Oacc[t][j] / lrow[j];
      qmem_out[(size_t)gm * DIM + h * DH + gn] = (bf16)o;
    }
}

// ================= elementwise epilogues =================
__global__ void k_silu_poly(const float* __restrict__ in, bf16* __restrict__ out, size_t n) {
  for (size_t i = (size_t)blockIdx.x * blockDim.x + threadIdx.x; i < n;
       i += (size_t)gridDim.x * blockDim.x) {
    float p = siluf_(in[i]);
    out[i] = (bf16)(p + p * p);
  }
}
__global__ void k_swiglu(const float* __restrict__ g1, const float* __restrict__ g3,
                         bf16* __restrict__ out, size_t n) {
  for (size_t i = (size_t)blockIdx.x * blockDim.x + threadIdx.x; i < n;
       i += (size_t)gridDim.x * blockDim.x)
    out[i] = (bf16)(siluf_(g1[i]) * g3[i]);
}
__global__ void k_gate_combine(const float* __restrict__ x, const float* __restrict__ attn,
                               const float* __restrict__ mem, float* __restrict__ x1, size_t n) {
  for (size_t i = (size_t)blockIdx.x * blockDim.x + threadIdx.x; i < n;
       i += (size_t)gridDim.x * blockDim.x)
    x1[i] = x[i] + attn[i] * sigmoidf_(mem[i]);
}
__global__ void k_final_add(const float* __restrict__ x1, const float* __restrict__ ffn,
                            float* __restrict__ out, size_t n) {
  for (size_t i = (size_t)blockIdx.x * blockDim.x + threadIdx.x; i < n;
       i += (size_t)gridDim.x * blockDim.x)
    out[i] = x1[i] + ffn[i];
}

// ================= launch =================
extern "C" void kernel_launch(void* const* d_in, const int* in_sizes, int n_in,
                              void* d_out, int out_size, void* d_ws, size_t ws_size,
                              hipStream_t stream) {
  const float* x       = (const float*)d_in[0];
  const float* w_qkv   = (const float*)d_in[1];
  const float* q_scale = (const float*)d_in[2];
  const float* k_scale = (const float*)d_in[3];
  const float* norm1_w = (const float*)d_in[4];
  const float* norm2_w = (const float*)d_in[5];
  const float* gg_w1   = (const float*)d_in[6];
  const float* gg_w2   = (const float*)d_in[7];
  const float* w_o     = (const float*)d_in[8];
  const float* mem_w1  = (const float*)d_in[9];
  const float* mem_w2  = (const float*)d_in[10];
  const float* ffn_w1  = (const float*)d_in[11];
  const float* ffn_w3  = (const float*)d_in[12];
  const float* ffn_w2  = (const float*)d_in[13];
  float* out = (float*)d_out;

  char* ws = (char*)d_ws;
  size_t off = 0;
  auto alloc = [&](size_t bytes) -> void* {
    void* p = ws + off;
    off += (bytes + 255) & ~(size_t)255;
    return p;
  };

  // bf16 weight buffers, all stored TRANSPOSED as [N][K] for TDM tile loads
  bf16* wqkv_t  = (bf16*)alloc((size_t)N3  * DIM * 2);   // [6144][2048]
  bf16* wo_t    = (bf16*)alloc((size_t)DIM * DIM * 2);   // [2048][2048]
  bf16* memw1_t = (bf16*)alloc((size_t)MH  * DIM * 2);   // [8192][2048]
  bf16* memw2_t = (bf16*)alloc((size_t)DIM * MH  * 2);   // [2048][8192]
  bf16* ffnw1_t = (bf16*)alloc((size_t)FHP * DIM * 2);   // [5504][2048]
  bf16* ffnw3_t = (bf16*)alloc((size_t)FHP * DIM * 2);   // [5504][2048]
  bf16* ffnw2_t = (bf16*)alloc((size_t)DIM * FHP * 2);   // [2048][5504]
  // bf16 activations
  bf16* h_bf     = (bf16*)alloc((size_t)SEQ * DIM * 2);
  bf16* q_bf     = (bf16*)alloc((size_t)SEQ * DIM * 2);
  bf16* k_bf     = (bf16*)alloc((size_t)SEQ * DIM * 2);
  bf16* v_bf     = (bf16*)alloc((size_t)SEQ * DIM * 2);
  bf16* attn_bf  = (bf16*)alloc((size_t)SEQ * DIM * 2);
  bf16* qmem_bf  = (bf16*)alloc((size_t)SEQ * DIM * 2);
  bf16* y_bf     = (bf16*)alloc((size_t)SEQ * DIM * 2);
  bf16* poly_bf  = (bf16*)alloc((size_t)SEQ * MH  * 2);
  bf16* act_bf   = (bf16*)alloc((size_t)SEQ * FHP * 2);
  // f32 buffers
  float* scrA      = (float*)alloc((size_t)SEQ * MH  * 4);  // qkv_f -> memh_f -> g1_f
  float* scrB      = (float*)alloc((size_t)SEQ * FHP * 4);  // g3_f
  float* gamma_f   = (float*)alloc((size_t)SEQ * 4);
  float* attnproj  = (float*)alloc((size_t)SEQ * DIM * 4);  // also reused as ffnout
  float* memout_f  = (float*)alloc((size_t)SEQ * DIM * 4);
  float* x1_f      = (float*)alloc((size_t)SEQ * DIM * 4);

  const int TB = 256;
  auto gridFor = [](size_t n, int tb) { return (unsigned)((n + tb - 1) / tb); };

  // 1) weights fp32 -> bf16, transposed to [N][K] (FFN hidden padded 5461 -> 5504)
  k_cast_pad_t<<<gridFor((size_t)N3 * DIM, TB),  TB, 0, stream>>>(w_qkv,  wqkv_t,  DIM, N3,  N3,  DIM);
  k_cast_pad_t<<<gridFor((size_t)DIM * DIM, TB), TB, 0, stream>>>(w_o,    wo_t,    DIM, DIM, DIM, DIM);
  k_cast_pad_t<<<gridFor((size_t)MH * DIM, TB),  TB, 0, stream>>>(mem_w1, memw1_t, DIM, MH,  MH,  DIM);
  k_cast_pad_t<<<gridFor((size_t)DIM * MH, TB),  TB, 0, stream>>>(mem_w2, memw2_t, MH,  DIM, DIM, MH);
  k_cast_pad_t<<<gridFor((size_t)FHP * DIM, TB), TB, 0, stream>>>(ffn_w1, ffnw1_t, DIM, FH,  FHP, DIM);
  k_cast_pad_t<<<gridFor((size_t)FHP * DIM, TB), TB, 0, stream>>>(ffn_w3, ffnw3_t, DIM, FH,  FHP, DIM);
  k_cast_pad_t<<<gridFor((size_t)DIM * FHP, TB), TB, 0, stream>>>(ffn_w2, ffnw2_t, FH,  DIM, DIM, FHP);

  // 2) h = rmsnorm(x, norm1)
  k_rmsnorm_bf16<<<SEQ, 256, 0, stream>>>(x, norm1_w, h_bf);
  // 3) gamma gate
  k_gamma<<<SEQ, 64, 0, stream>>>(h_bf, gg_w1, gg_w2, gamma_f);
  // 4) qkv = h @ w_qkv
  {
    dim3 g(N3 / BN, SEQ / BM);
    k_gemm_tdm<<<g, TB, 0, stream>>>(h_bf, wqkv_t, scrA, SEQ, N3, DIM);
  }
  // 5) per-head rmsnorm + rope, split heads
  {
    dim3 g(SEQ, NH);
    k_qknorm_rope<<<g, DH, 0, stream>>>(scrA, q_scale, k_scale, q_bf, k_bf, v_bf);
  }
  // 6) sliding-window attention
  {
    dim3 g(SEQ / 64, NH);
    k_swa<<<g, 128, 0, stream>>>(q_bf, k_bf, v_bf, attn_bf);
  }
  // 7) omega memory
  {
    dim3 g(SEQ / 64, NH);
    k_omega<<<g, 128, 0, stream>>>(q_bf, k_bf, gamma_f, qmem_bf);
  }
  // 8) attn projection
  {
    dim3 g(DIM / BN, SEQ / BM);
    k_gemm_tdm<<<g, TB, 0, stream>>>(attn_bf, wo_t, attnproj, SEQ, DIM, DIM);
  }
  // 9) mem hidden
  {
    dim3 g(MH / BN, SEQ / BM);
    k_gemm_tdm<<<g, TB, 0, stream>>>(qmem_bf, memw1_t, scrA, SEQ, MH, DIM);
  }
  // 10) poly = silu + silu^2
  k_silu_poly<<<gridFor((size_t)SEQ * MH, TB), TB, 0, stream>>>(scrA, poly_bf, (size_t)SEQ * MH);
  // 11) mem out
  {
    dim3 g(DIM / BN, SEQ / BM);
    k_gemm_tdm<<<g, TB, 0, stream>>>(poly_bf, memw2_t, memout_f, SEQ, DIM, MH);
  }
  // 12) x1 = x + attnproj * sigmoid(memout)
  k_gate_combine<<<gridFor((size_t)SEQ * DIM, TB), TB, 0, stream>>>(x, attnproj, memout_f, x1_f,
                                                                   (size_t)SEQ * DIM);
  // 13) y = rmsnorm(x1, norm2)
  k_rmsnorm_bf16<<<SEQ, 256, 0, stream>>>(x1_f, norm2_w, y_bf);
  // 14/15) g1 = y@w1, g3 = y@w3
  {
    dim3 g(FHP / BN, SEQ / BM);
    k_gemm_tdm<<<g, TB, 0, stream>>>(y_bf, ffnw1_t, scrA, SEQ, FHP, DIM);
    k_gemm_tdm<<<g, TB, 0, stream>>>(y_bf, ffnw3_t, scrB, SEQ, FHP, DIM);
  }
  // 16) act = silu(g1)*g3
  k_swiglu<<<gridFor((size_t)SEQ * FHP, TB), TB, 0, stream>>>(scrA, scrB, act_bf, (size_t)SEQ * FHP);
  // 17) ffn out (reuse attnproj buffer)
  {
    dim3 g(DIM / BN, SEQ / BM);
    k_gemm_tdm<<<g, TB, 0, stream>>>(act_bf, ffnw2_t, attnproj, SEQ, DIM, FHP);
  }
  // 18) out = x1 + ffnout
  k_final_add<<<gridFor((size_t)SEQ * DIM, TB), TB, 0, stream>>>(x1_f, attnproj, out,
                                                                 (size_t)SEQ * DIM);
}
